// EncoderLayer_53996328846043
// MI455X (gfx1250) — matrix operand order repaired and checked by reading, not compiled, not run on hardware
//
#include <hip/hip_runtime.h>
#include <hip/hip_bf16.h>

// ---------------------------------------------------------------------------
// Types for CDNA5 WMMA (wave32): D(f32 16x16) = A(bf16 16x32) * B(bf16 32x16) + C
// ---------------------------------------------------------------------------
typedef __attribute__((ext_vector_type(16))) __bf16 v16bf;
typedef __attribute__((ext_vector_type(8)))  __bf16 v8bf;
typedef __attribute__((ext_vector_type(8)))  float  v8f;

__device__ __forceinline__ v8f wmma_bf16(v16bf a, v16bf b, v8f c) {
  return __builtin_amdgcn_wmma_f32_16x16x32_bf16(false, a, false, b, (short)0, c,
                                                 false, false);
}

__device__ __forceinline__ v16bf cat8(v8bf lo, v8bf hi) {
  union { v16bf v; v8bf h[2]; } u;
  u.h[0] = lo; u.h[1] = hi;
  return u.v;
}

// CDNA5 async DMA: 16 bytes global -> LDS, tracked with ASYNCcnt.
// lds_off = wave-relative LDS byte offset (low 32 bits of generic address).
__device__ __forceinline__ void async_copy_b128(unsigned lds_off,
                                                const __bf16* gptr) {
  asm volatile("global_load_async_to_lds_b128 %0, %1, off"
               :: "v"(lds_off), "v"(gptr) : "memory");
}
__device__ __forceinline__ void wait_asynccnt0() {
  asm volatile("s_wait_asynccnt 0" ::: "memory");
}
__device__ __forceinline__ unsigned lds_off32(const void* p) {
  return (unsigned)(unsigned long long)p;   // LDS address is the low 32 bits of the pointer
}

// ---------------------------------------------------------------------------
// Generic bf16 GEMM:  C[M,N] = A[M,K] @ Wt[N,K]^T + bias  (optional ReLU)
// Block tile 128x128, 8 waves (2x4), wave tile 64x32, K step 32.
// A tile DMA'd into double-buffered LDS with global_load_async_to_lds_b128;
// B fragments streamed from L2-resident weights.
// ---------------------------------------------------------------------------
template <int RELU, int OUTBF>
__global__ __launch_bounds__(256) void gemm_bf16_kernel(
    const __bf16* __restrict__ A, const __bf16* __restrict__ Wt,
    const float* __restrict__ bias, float* __restrict__ Cf,
    __bf16* __restrict__ Cb, int M, int N, int K) {
  const int LDA = 40;                       // 32 + 8 pad (avoids bank conflicts)
  __shared__ __bf16 sA[2][128 * LDA];

  const int tid = threadIdx.x;
  const int lane = tid & 31, wid = tid >> 5;
  const int waveM = wid & 1, waveN = wid >> 1;   // 2 x 4 wave grid
  const int l16 = lane & 15, lh = lane >> 4;
  const int bm = blockIdx.y * 128;
  const int bn = blockIdx.x * 128;

  v8f acc[4][2] = {};

  // cooperative A-tile DMA coords: 2 threads per row, 16 elems (32 B) each
  const int arow = tid >> 1;
  const int acol = (tid & 1) * 16;
  const __bf16* Ag = A + (long)(bm + arow) * K + acol;
  const unsigned ldsw0 = lds_off32(&sA[0][arow * LDA + acol]);
  const unsigned ldsw1 = lds_off32(&sA[1][arow * LDA + acol]);

  const __bf16* Bb0 = Wt + (long)(bn + waveN * 32 + l16) * K + lh * 8;
  const __bf16* Bb1 = Bb0 + (long)16 * K;
  const __bf16* sAr0 = &sA[0][(waveM * 64 + l16) * LDA + lh * 8];
  const __bf16* sAr1 = &sA[1][(waveM * 64 + l16) * LDA + lh * 8];

  const int nk = K >> 5;
  // prologue: DMA tile 0 into buffer 0
  async_copy_b128(ldsw0, Ag);
  async_copy_b128(ldsw0 + 16, Ag + 8);

  for (int i = 0; i < nk; i++) {
    wait_asynccnt0();          // this wave's DMA portion of tile i landed
    __syncthreads();           // ...and everyone else's

    if (i + 1 < nk) {          // overlap: DMA tile i+1 while computing tile i
      const __bf16* g = Ag + (i + 1) * 32;
      const unsigned d = ((i + 1) & 1) ? ldsw1 : ldsw0;
      async_copy_b128(d, g);
      async_copy_b128(d + 16, g + 8);
    }

    const int k0 = i * 32;
    v16bf b0 = cat8(*(const v8bf*)(Bb0 + k0), *(const v8bf*)(Bb0 + k0 + 16));
    v16bf b1 = cat8(*(const v8bf*)(Bb1 + k0), *(const v8bf*)(Bb1 + k0 + 16));
    const __bf16* sAr = (i & 1) ? sAr1 : sAr0;
#pragma unroll
    for (int mt = 0; mt < 4; mt++) {
      const __bf16* pa = sAr + mt * 16 * LDA;
      v16bf a = cat8(*(const v8bf*)pa, *(const v8bf*)(pa + 16));
      acc[mt][0] = wmma_bf16(a, b0, acc[mt][0]);
      acc[mt][1] = wmma_bf16(a, b1, acc[mt][1]);
    }
    __syncthreads();           // all reads of buffer i&1 done before reuse
  }

  // Epilogue.  C layout: VGPR v -> row v + 8*(lane>=16), col = lane&15.
  const int crow0 = bm + waveM * 64 + lh * 8;
  const int ccol0 = bn + waveN * 32 + l16;
#pragma unroll
  for (int nt = 0; nt < 2; nt++) {
    int col = ccol0 + nt * 16;
    float bval = bias[col];
#pragma unroll
    for (int mt = 0; mt < 4; mt++) {
#pragma unroll
      for (int v = 0; v < 8; v++) {
        long r = (long)(crow0 + mt * 16 + v);
        float val = acc[mt][nt][v] + bval;
        if (RELU) val = fmaxf(val, 0.0f);
        if (OUTBF) Cb[r * N + col] = (__bf16)val;
        else       Cf[r * N + col] = val;
      }
    }
  }
}

// ---------------------------------------------------------------------------
// Chunked local attention.  One block = (b, chunk c, head h, 16-query block).
// Keys for chunk c are tokens [(c-1)*W, (c+1)*W); first W of chunk 0 masked.
// qkv: [B*L, 3D] bf16 (cols: q | k | v).  o: [B*L, D] bf16.
// ---------------------------------------------------------------------------
__global__ __launch_bounds__(256) void attn_kernel(
    const __bf16* __restrict__ qkv, __bf16* __restrict__ o) {
  const int Dm = 1024, HD = 64, LD3 = 3072, Lc = 4096, Wc = 256;
  __shared__ float  sS[16 * 512];    // scores; later reused as partial O [8][16*64]
  __shared__ __bf16 sP[16 * 512];    // probabilities (unnormalized exp)
  __shared__ float  sRed[16 * 16];
  __shared__ float  sMax[16];
  __shared__ float  sSum[16];

  int idx = blockIdx.x;
  const int qb = idx & 15; idx >>= 4;
  const int h  = idx & 15; idx >>= 4;
  const int c  = idx & 15; idx >>= 4;
  const int b  = idx;

  const int tid = threadIdx.x, lane = tid & 31, wid = tid >> 5;
  const int l16 = lane & 15, lh = lane >> 4;

  const long qrow   = (long)b * Lc + c * Wc + qb * 16;  // first query token row
  const long kvbase = (long)b * Lc;
  const int  kwin0  = (c - 1) * Wc;                     // key j -> token kwin0 + j

  // ---- Q fragments: A operand 16x64 = two 16x32 frags (global, K-contig rows)
  const __bf16* qp = qkv + (qrow + l16) * LD3 + h * HD + lh * 8;
  v16bf aq0 = cat8(*(const v8bf*)(qp +  0), *(const v8bf*)(qp + 16));
  v16bf aq1 = cat8(*(const v8bf*)(qp + 32), *(const v8bf*)(qp + 48));

  // ---- scores: wave `wid` owns keys [wid*64, wid*64+64)
  const float scale = 0.125f;  // 1/sqrt(64)
  const int kb = wid * 64;
#pragma unroll
  for (int nt = 0; nt < 4; nt++) {
    const int key  = kb + nt * 16 + l16;    // this lane's key column
    const int tok  = kwin0 + key;
    const int tokc = tok < 0 ? 0 : tok;
    const __bf16* kp = qkv + (kvbase + tokc) * LD3 + Dm + h * HD + lh * 8;
    v16bf bk0 = cat8(*(const v8bf*)(kp +  0), *(const v8bf*)(kp + 16));
    v16bf bk1 = cat8(*(const v8bf*)(kp + 32), *(const v8bf*)(kp + 48));
    v8f s = {};
    s = wmma_bf16(aq0, bk0, s);
    s = wmma_bf16(aq1, bk1, s);
#pragma unroll
    for (int v = 0; v < 8; v++) {
      int r = v + lh * 8;
      sS[r * 512 + kb + nt * 16 + l16] = (tok < 0) ? -1e30f : s[v] * scale;
    }
  }
  __syncthreads();

  // ---- softmax over 512 keys: 16 threads per row, interleaved columns
  const int r = tid >> 4, si = tid & 15;
  float m = -1e30f;
  for (int i = 0; i < 32; i++) m = fmaxf(m, sS[r * 512 + si + 16 * i]);
  sRed[r * 16 + si] = m;
  __syncthreads();
  if (si == 0) {
    float mm = -1e30f;
    for (int i = 0; i < 16; i++) mm = fmaxf(mm, sRed[r * 16 + i]);
    sMax[r] = mm;
  }
  __syncthreads();
  const float mm = sMax[r];
  float psum = 0.0f;
  for (int i = 0; i < 32; i++) {
    int j = si + 16 * i;
    float e = __expf(sS[r * 512 + j] - mm);
    psum += e;
    sP[r * 512 + j] = (__bf16)e;
  }
  sRed[r * 16 + si] = psum;
  __syncthreads();
  if (si == 0) {
    float t = 0.0f;
    for (int i = 0; i < 16; i++) t += sRed[r * 16 + i];
    sSum[r] = t;
  }
  __syncthreads();

  // ---- P @ V: wave accumulates a partial 16x64 O over its 64 keys
  v8f oacc[4] = {};
#pragma unroll
  for (int ks = 0; ks < 2; ks++) {
    const int k0 = kb + ks * 32;
    const __bf16* pp = sP + l16 * 512 + k0 + lh * 8;
    v16bf ap = cat8(*(const v8bf*)pp, *(const v8bf*)(pp + 16));
#pragma unroll
    for (int nt = 0; nt < 4; nt++) {
      const int hdcol = h * HD + nt * 16 + l16;
      v16bf bv;
#pragma unroll
      for (int i = 0; i < 16; i++) {
        int tok  = kwin0 + k0 + lh * 8 + (i < 8 ? i : i + 8);
        int tokc = tok < 0 ? 0 : tok;
        bv[i] = qkv[(kvbase + tokc) * LD3 + 2 * Dm + hdcol];
      }
      oacc[nt] = wmma_bf16(ap, bv, oacc[nt]);
    }
  }

  // ---- cross-wave O reduction through LDS (alias sS), normalize, store bf16
  float* sO = sS;
#pragma unroll
  for (int nt = 0; nt < 4; nt++)
#pragma unroll
    for (int v = 0; v < 8; v++)
      sO[(wid * 16 + v + lh * 8) * 64 + nt * 16 + l16] = oacc[nt][v];
  __syncthreads();

  for (int t = tid; t < 16 * 64; t += 256) {
    int rr = t >> 6, cc = t & 63;
    float a = 0.0f;
#pragma unroll
    for (int w = 0; w < 8; w++) a += sO[(w * 16 + rr) * 64 + cc];
    a /= sSum[rr];
    o[(qrow + rr) * Dm + h * HD + cc] = (__bf16)a;
  }
}

// ---------------------------------------------------------------------------
// residual add + LayerNorm (one block per row of 1024)
// ---------------------------------------------------------------------------
__global__ __launch_bounds__(256) void add_ln_kernel(
    const float* __restrict__ xa, const float* __restrict__ xb,
    const float* __restrict__ gamma, const float* __restrict__ beta,
    float* __restrict__ outf, __bf16* __restrict__ outb) {
  __shared__ float red[256];
  const long row = blockIdx.x;
  const int tid = threadIdx.x;
  float4 a = ((const float4*)(xa + row * 1024))[tid];
  float4 bb = ((const float4*)(xb + row * 1024))[tid];
  float rv[4] = {a.x + bb.x, a.y + bb.y, a.z + bb.z, a.w + bb.w};

  float s = rv[0] + rv[1] + rv[2] + rv[3];
  red[tid] = s;
  __syncthreads();
  for (int off = 128; off > 0; off >>= 1) {
    if (tid < off) red[tid] += red[tid + off];
    __syncthreads();
  }
  const float mean = red[0] * (1.0f / 1024.0f);
  __syncthreads();

  float vs = 0.0f;
#pragma unroll
  for (int k = 0; k < 4; k++) { float d = rv[k] - mean; vs += d * d; }
  red[tid] = vs;
  __syncthreads();
  for (int off = 128; off > 0; off >>= 1) {
    if (tid < off) red[tid] += red[tid + off];
    __syncthreads();
  }
  const float inv = rsqrtf(red[0] * (1.0f / 1024.0f) + 1e-6f);

  const int c0 = tid * 4;
#pragma unroll
  for (int k = 0; k < 4; k++) {
    float val = gamma[c0 + k] * (rv[k] - mean) * inv + beta[c0 + k];
    outf[row * 1024 + c0 + k] = val;
    if (outb) outb[row * 1024 + c0 + k] = (__bf16)val;
  }
}

// ---------------------------------------------------------------------------
// f32 -> bf16 conversion (8 elems/thread), and transpose-convert for w1/w2
// ---------------------------------------------------------------------------
__global__ void cvt_kernel(const float* __restrict__ in, __bf16* __restrict__ out,
                           long n) {
  long i = ((long)blockIdx.x * blockDim.x + threadIdx.x) * 8;
  if (i + 8 <= n) {
    float4 f0 = *(const float4*)(in + i);
    float4 f1 = *(const float4*)(in + i + 4);
    v8bf r;
    r[0] = (__bf16)f0.x; r[1] = (__bf16)f0.y; r[2] = (__bf16)f0.z; r[3] = (__bf16)f0.w;
    r[4] = (__bf16)f1.x; r[5] = (__bf16)f1.y; r[6] = (__bf16)f1.z; r[7] = (__bf16)f1.w;
    *(v8bf*)(out + i) = r;
  }
}

__global__ __launch_bounds__(256) void tcvt_kernel(
    const float* __restrict__ in, __bf16* __restrict__ out, int R, int C) {
  __shared__ float t[32][33];
  const int bx = blockIdx.x * 32;  // column tile of input
  const int by = blockIdx.y * 32;  // row tile of input
  const int tx = threadIdx.x & 31, ty = threadIdx.x >> 5;
#pragma unroll
  for (int i = 0; i < 4; i++)
    t[ty + i * 8][tx] = in[(long)(by + ty + i * 8) * C + bx + tx];
  __syncthreads();
#pragma unroll
  for (int i = 0; i < 4; i++)
    out[(long)(bx + ty + i * 8) * R + by + tx] = (__bf16)t[tx][ty + i * 8];
}

// ---------------------------------------------------------------------------
// Orchestration
// ---------------------------------------------------------------------------
extern "C" void kernel_launch(void* const* d_in, const int* in_sizes, int n_in,
                              void* d_out, int out_size, void* d_ws, size_t ws_size,
                              hipStream_t stream) {
  (void)in_sizes; (void)n_in; (void)out_size; (void)ws_size;
  const float* x         = (const float*)d_in[0];
  const float* in_proj_w = (const float*)d_in[1];
  const float* in_proj_b = (const float*)d_in[2];
  const float* out_w     = (const float*)d_in[3];
  const float* out_b     = (const float*)d_in[4];
  const float* gamma1    = (const float*)d_in[5];
  const float* beta1     = (const float*)d_in[6];
  const float* w1        = (const float*)d_in[7];
  const float* bf1       = (const float*)d_in[8];
  const float* w2        = (const float*)d_in[9];
  const float* bf2       = (const float*)d_in[10];
  const float* gamma2    = (const float*)d_in[11];
  const float* beta2     = (const float*)d_in[12];
  float* out = (float*)d_out;

  const long MBy = 1024L * 1024L;
  char* ws = (char*)d_ws;
  __bf16* xb    = (__bf16*)(ws + 0 * MBy);     // 32 MB  [16384,1024]
  __bf16* wqkv  = (__bf16*)(ws + 32 * MBy);    // 6 MB   [3072,1024]
  __bf16* woutb = (__bf16*)(ws + 38 * MBy);    // 2 MB   [1024,1024]
  __bf16* w1t   = (__bf16*)(ws + 40 * MBy);    // 8 MB   [4096,1024]
  __bf16* w2t   = (__bf16*)(ws + 48 * MBy);    // 8 MB   [1024,4096]
  __bf16* qkvb  = (__bf16*)(ws + 56 * MBy);    // 96 MB  [16384,3072]
  __bf16* ob    = (__bf16*)(ws + 152 * MBy);   // 32 MB  [16384,1024]
  float*  attnf = (float*)(ws + 184 * MBy);    // 64 MB  [16384,1024]
  float*  x1f   = (float*)(ws + 248 * MBy);    // 64 MB
  __bf16* x1b   = (__bf16*)(ws + 312 * MBy);   // 32 MB
  __bf16* hb    = (__bf16*)(ws + 56 * MBy);    // 128 MB, aliases qkvb+ob (both dead)
  float*  ffnf  = (float*)(ws + 184 * MBy);    // aliases attnf (dead after ln1)

  const int M = 16384;  // B*L

  // precision conversions (weights L2-resident thereafter)
  cvt_kernel<<<8192, 256, 0, stream>>>(x, xb, (long)M * 1024);
  cvt_kernel<<<1536, 256, 0, stream>>>(in_proj_w, wqkv, 3072L * 1024L);
  cvt_kernel<<<512, 256, 0, stream>>>(out_w, woutb, 1024L * 1024L);
  tcvt_kernel<<<dim3(128, 32), 256, 0, stream>>>(w1, w1t, 1024, 4096);  // -> [4096,1024]
  tcvt_kernel<<<dim3(32, 128), 256, 0, stream>>>(w2, w2t, 4096, 1024);  // -> [1024,4096]

  // QKV projection: [16384,1024] x [3072,1024]^T -> bf16 [16384,3072]
  gemm_bf16_kernel<0, 1><<<dim3(24, 128), 256, 0, stream>>>(
      xb, wqkv, in_proj_b, nullptr, qkvb, M, 3072, 1024);

  // chunked local attention -> bf16 [16384,1024]
  attn_kernel<<<16384, 256, 0, stream>>>(qkvb, ob);

  // output projection -> f32
  gemm_bf16_kernel<0, 0><<<dim3(8, 128), 256, 0, stream>>>(
      ob, woutb, out_b, attnf, nullptr, M, 1024, 1024);

  // residual + LN1 -> x1 (f32 for residual, bf16 for FFN GEMM)
  add_ln_kernel<<<M, 256, 0, stream>>>(x, attnf, gamma1, beta1, x1f, x1b);

  // FFN: relu(x1 @ w1 + bf1) -> bf16 [16384,4096]
  gemm_bf16_kernel<1, 1><<<dim3(32, 128), 256, 0, stream>>>(
      x1b, w1t, bf1, nullptr, hb, M, 4096, 1024);

  // FFN second matmul -> f32
  gemm_bf16_kernel<0, 0><<<dim3(8, 128), 256, 0, stream>>>(
      hb, w2t, bf2, ffnf, nullptr, M, 1024, 4096);

  // residual + LN2 -> output
  add_ln_kernel<<<M, 256, 0, stream>>>(x1f, ffnf, gamma2, beta2, out, nullptr);
}
